// MMHA_78039555768536
// MI455X (gfx1250) — matrix operand order repaired and checked by reading, not compiled, not run on hardware
//
#include <hip/hip_runtime.h>
#include <cstdint>

// ---------------------------------------------------------------------------
// Gated multi-head attention + linear-attention memory (delta rule), CDNA5.
// B=4, L=8192, DIN=HD=512, H=8, D=64, SEG=512, NSEG=16.
//
// Phase 1 (parallel):  Q/K/V = X@W*  (bf16 WMMA, W tiles staged via TDM)
// Phase 2 (parallel):  causal flash attention, K/V tiles staged in LDS
// Phase 3 (parallel):  z prefix per segment; denominators sq.z / sk.z
// Phase 4 (scan x16):  read  R=[sq;sk]@M (M tile via TDM) ;
//                      update M += sk^T @ delta (LDS-staged bf16 tiles)
// Phase 5 (parallel):  out = out_pre @ Wd (Wd tile via TDM)
// ---------------------------------------------------------------------------

typedef __attribute__((ext_vector_type(16))) __bf16 v16bf;
typedef __attribute__((ext_vector_type(8)))  float  v8f;
typedef unsigned int u32x4 __attribute__((ext_vector_type(4)));
typedef int          i32x4 __attribute__((ext_vector_type(4)));
typedef int          i32x8 __attribute__((ext_vector_type(8)));

#define LANE (threadIdx.x & 31)

#if defined(__HIP_DEVICE_COMPILE__) && __has_builtin(__builtin_amdgcn_tensor_load_to_lds)
#define HAVE_TDM 1
#else
#define HAVE_TDM 0
#endif

__device__ __forceinline__ v8f wmma_bf16(v16bf a, v16bf b, v8f c) {
  return __builtin_amdgcn_wmma_f32_16x16x32_bf16(false, a, false, b, (short)0, c,
                                                 false, false);
}

__device__ __forceinline__ float elu1f(float x) {
  return x > 0.f ? x + 1.f : __expf(x);
}

// in-lane K offset for A-matrix (16x32, 16-bit)
__device__ __forceinline__ int a_koff(int p, int h) {
  return (p < 4) ? (2 * p + 8 * h) : (16 + 2 * (p - 4) + 8 * h);
}
// in-lane K offset for B-matrix (32x16, 16-bit)
__device__ __forceinline__ int b_koff(int p, int h) { return (p < 4) ? (2 * p + 8 * h) : (16 + 2 * (p - 4) + 8 * h); }

// A frag (M=16,K=32) from row-major f32 [m][k]
__device__ __forceinline__ v16bf a_frag_f32(const float* S, int ld, int m0, int kb) {
  int l = LANE, m = m0 + (l & 15), h = l >> 4;
  const float* row = S + (size_t)m * ld + kb;
  v16bf f;
#pragma unroll
  for (int p = 0; p < 8; ++p) {
    int k = a_koff(p, h);
    f[2 * p]     = (__bf16)row[k];
    f[2 * p + 1] = (__bf16)row[k + 1];
  }
  return f;
}

// A frag from row-major bf16 [m][k]
__device__ __forceinline__ v16bf a_frag_bf16(const __bf16* S, int ld, int m0, int kb) {
  int l = LANE, m = m0 + (l & 15), h = l >> 4;
  const __bf16* row = S + (size_t)m * ld + kb;
  v16bf f;
#pragma unroll
  for (int p = 0; p < 8; ++p) {
    int k = a_koff(p, h);
    f[2 * p]     = row[k];
    f[2 * p + 1] = row[k + 1];
  }
  return f;
}

// B frag (K=32,N=16) from row-major f32 [k][n]
__device__ __forceinline__ v16bf b_frag_f32(const float* S, int ld, int kb, int n0) {
  int l = LANE, n = n0 + (l & 15), h = l >> 4;
  v16bf f;
#pragma unroll
  for (int p = 0; p < 8; ++p) {
    int k = kb + b_koff(p, h);
    f[2 * p]     = (__bf16)S[(size_t)k * ld + n];
    f[2 * p + 1] = (__bf16)S[(size_t)(k + 1) * ld + n];
  }
  return f;
}

// B frag from row-major bf16 [k][n]
__device__ __forceinline__ v16bf b_frag_bf16(const __bf16* S, int ld, int kb, int n0) {
  int l = LANE, n = n0 + (l & 15), h = l >> 4;
  v16bf f;
#pragma unroll
  for (int p = 0; p < 8; ++p) {
    int k = kb + b_koff(p, h);
    f[2 * p]     = S[(size_t)k * ld + n];
    f[2 * p + 1] = S[(size_t)(k + 1) * ld + n];
  }
  return f;
}

// B frag from TRANSPOSED bf16 source: B[k][n] = S[n][k]  (K^T in attention)
__device__ __forceinline__ v16bf bT_frag_bf16(const __bf16* S, int ld, int n0, int kb) {
  int l = LANE, n = n0 + (l & 15), h = l >> 4;
  const __bf16* row = S + (size_t)n * ld + kb;
  v16bf f;
#pragma unroll
  for (int p = 0; p < 8; ++p) {
    int k = b_koff(p, h);
    f[2 * p]     = row[k];
    f[2 * p + 1] = row[k + 1];
  }
  return f;
}

// A frag over segment-local rows m in [0,2048): row m -> global row b*8192+t*512+s
__device__ __forceinline__ v16bf a_frag_seg(const __bf16* S, int t, int m0, int kb) {
  int l = LANE, m = m0 + (l & 15), h = l >> 4;
  size_t grow = (size_t)(m >> 9) * 8192 + (size_t)t * 512 + (m & 511);
  const __bf16* row = S + grow * 512 + kb;
  v16bf f;
#pragma unroll
  for (int p = 0; p < 8; ++p) {
    int k = a_koff(p, h);
    f[2 * p]     = row[k];
    f[2 * p + 1] = row[k + 1];
  }
  return f;
}

// A frag of sk^T from LDS tile At[32 kk][64 m]: A[m][kk] = At[kk][m]
__device__ __forceinline__ v16bf aT_frag_lds(const __bf16* S, int m0) {
  int l = LANE, m = m0 + (l & 15), h = l >> 4;
  v16bf f;
#pragma unroll
  for (int p = 0; p < 8; ++p) {
    int kk = a_koff(p, h);
    f[2 * p]     = S[kk * 64 + m];
    f[2 * p + 1] = S[(kk + 1) * 64 + m];
  }
  return f;
}

// ---------------------------------------------------------------------------
// Stage a tile of f32 (th rows x tw cols, source row stride ld elems) into
// LDS, compact row-major. TDM path: one descriptor from wave 0, completion
// via TENSORcnt. Fallback: cooperative float4 copy.
// ---------------------------------------------------------------------------
__device__ __forceinline__ void stage_tile_f32(float* lds, const float* g, int ld,
                                               int tw, int th) {
#if HAVE_TDM
  if (threadIdx.x < 32) {
    unsigned lds_off = (unsigned)(uintptr_t)lds;
    unsigned long long ga = (unsigned long long)(uintptr_t)g;
    u32x4 g0 = {};
    g0[0] = 1u;                                              // count=1 (valid D#)
    g0[1] = lds_off;                                         // lds_addr
    g0[2] = (unsigned)ga;                                    // global_addr[31:0]
    g0[3] = (unsigned)((ga >> 32) & 0x01FFFFFFu) | (2u << 30);  // addr[56:32]|type=2
    i32x8 g1 = {};
    g1[0] = (int)(2u << 16);                                 // data_size=4B, mask=0
    g1[1] = (int)(((unsigned)ld & 0xFFFFu) << 16);           // tensor_dim0 lo16
    g1[2] = (int)(((unsigned)ld >> 16) | (((unsigned)th & 0xFFFFu) << 16));
    g1[3] = (int)((((unsigned)th >> 16) & 0xFFFFu) | (((unsigned)tw & 0xFFFFu) << 16));
    g1[4] = (int)((unsigned)th & 0xFFFFu);                   // tile_dim1, tile_dim2=0
    g1[5] = (int)(unsigned)ld;                               // dim0_stride lo32
    g1[6] = 0;
    g1[7] = 0;
    i32x4 z4 = {};
#if __clang_major__ >= 23
    i32x8 z8 = {};
    __builtin_amdgcn_tensor_load_to_lds(g0, g1, z4, z4, z8, 0);
#else
    __builtin_amdgcn_tensor_load_to_lds(g0, g1, z4, z4, 0);
#endif
    __builtin_amdgcn_s_wait_tensorcnt(0);
  }
  __syncthreads();
#else
  int n4 = (tw * th) >> 2;
  for (int i = threadIdx.x; i < n4; i += blockDim.x) {
    int e = i << 2, r = e / tw, c = e % tw;
    *(float4*)(lds + e) = *(const float4*)(g + (size_t)r * ld + c);
  }
  __syncthreads();
#endif
}

// ---------------------------------------------------------------------------
// Phase 1: QKV projection  (M=32768, N=512, K=512), z-dim selects Wq/Wk/Wv
// ---------------------------------------------------------------------------
__global__ void proj_kernel(const float* __restrict__ X, const float* __restrict__ Wq,
                            const float* __restrict__ Wk, const float* __restrict__ Wv,
                            __bf16* Qb, __bf16* Kb, __bf16* Vb, __bf16* sqb, __bf16* skb) {
  __shared__ __align__(16) float Wt[32 * 128];
  int which = blockIdx.z;
  const float* W = (which == 0) ? Wq : (which == 1) ? Wk : Wv;
  int wave = threadIdx.x >> 5, wm = wave >> 2, wn = wave & 3;
  int m0 = blockIdx.x * 64 + wm * 32;
  int nblk = blockIdx.y * 128, n0l = wn * 32;
  v8f acc[2][2] = {};
  for (int k = 0; k < 512; k += 32) {
    stage_tile_f32(Wt, W + (size_t)k * 512 + nblk, 512, 128, 32);
    __builtin_prefetch(X + (size_t)m0 * 512 + k + 64, 0, 0);
    v16bf a0 = a_frag_f32(X, 512, m0, k);
    v16bf a1 = a_frag_f32(X, 512, m0 + 16, k);
    v16bf b0 = b_frag_f32(Wt, 128, 0, n0l);
    v16bf b1 = b_frag_f32(Wt, 128, 0, n0l + 16);
    acc[0][0] = wmma_bf16(a0, b0, acc[0][0]);
    acc[0][1] = wmma_bf16(a0, b1, acc[0][1]);
    acc[1][0] = wmma_bf16(a1, b0, acc[1][0]);
    acc[1][1] = wmma_bf16(a1, b1, acc[1][1]);
    __syncthreads();
  }
  int l = LANE, h = l >> 4, nc = l & 15;
#pragma unroll
  for (int i = 0; i < 2; ++i)
#pragma unroll
    for (int j = 0; j < 2; ++j)
#pragma unroll
      for (int r = 0; r < 8; ++r) {
        size_t m = m0 + i * 16 + r + 8 * h;
        size_t idx = m * 512 + (nblk + n0l + j * 16 + nc);
        float e = acc[i][j][r];
        if (which == 0) { Qb[idx] = (__bf16)e; sqb[idx] = (__bf16)elu1f(e); }
        else if (which == 1) { Kb[idx] = (__bf16)e; skb[idx] = (__bf16)elu1f(e); }
        else { Vb[idx] = (__bf16)e; }
      }
}

// ---------------------------------------------------------------------------
// Phase 2: flash-style causal attention per (seg,b,h); writes (1-g)*attn.
// grid (qtile=4, seg=16, b*h=32), 256 threads (8 waves x 16 q-rows).
// K/V tiles (32 keys x 64 d) staged in LDS once per block.
// ---------------------------------------------------------------------------
__global__ void attn_kernel(const __bf16* __restrict__ Qb, const __bf16* __restrict__ Kb,
                            const __bf16* __restrict__ Vb, const float* __restrict__ beta,
                            float* __restrict__ out_pre) {
  __shared__ __align__(16) __bf16 Pld[8][16 * 32];
  __shared__ __align__(16) __bf16 Kt[32 * 64];
  __shared__ __align__(16) __bf16 Vt[32 * 64];
  int qt = blockIdx.x, seg = blockIdx.y;
  int b = blockIdx.z >> 3, hh = blockIdx.z & 7;
  int wave = threadIdx.x >> 5;
  int l = LANE, half = l >> 4, nc = l & 15;
  float g = 1.f / (1.f + __expf(-beta[hh]));
  float og = 1.f - g;
  int m0l = qt * 128 + wave * 16;
  size_t rowbase = (size_t)b * 8192 + (size_t)seg * 512;
  const __bf16* Qp = Qb + rowbase * 512 + hh * 64;
  const __bf16* Kp = Kb + rowbase * 512 + hh * 64;
  const __bf16* Vp = Vb + rowbase * 512 + hh * 64;

  v16bf aq0 = a_frag_bf16(Qp, 512, m0l, 0);
  v16bf aq1 = a_frag_bf16(Qp, 512, m0l, 32);
  float mi[8], li[8];
  v8f o[4] = {};
#pragma unroll
  for (int r = 0; r < 8; ++r) { mi[r] = -1e30f; li[r] = 0.f; }

  const float inv = 0.125f;  // 1/sqrt(64)
  int my_ntiles = (m0l + 15) / 32 + 1;
  int blk_ntiles = qt * 4 + 4;
  for (int kt = 0; kt < blk_ntiles; ++kt) {
    int kb = kt * 32;
    {  // cooperative K/V tile stage: 32 rows x 64 bf16, one uint4 each per thread
      int row = threadIdx.x >> 3, c8 = (threadIdx.x & 7) * 8;
      *(uint4*)(Kt + row * 64 + c8) = *(const uint4*)(Kp + (size_t)(kb + row) * 512 + c8);
      *(uint4*)(Vt + row * 64 + c8) = *(const uint4*)(Vp + (size_t)(kb + row) * 512 + c8);
    }
    __syncthreads();
    if (kt < my_ntiles) {
      v8f s0 = {}, s1 = {};
      s0 = wmma_bf16(aq0, bT_frag_bf16(Kt, 64, 0, 0), s0);
      s0 = wmma_bf16(aq1, bT_frag_bf16(Kt, 64, 0, 32), s0);
      s1 = wmma_bf16(aq0, bT_frag_bf16(Kt, 64, 16, 0), s1);
      s1 = wmma_bf16(aq1, bT_frag_bf16(Kt, 64, 16, 32), s1);
#pragma unroll
      for (int r = 0; r < 8; ++r) {
        int qrow = m0l + r + 8 * half;
        float v0 = s0[r] * inv + ((kb + nc) > qrow ? -1e9f : 0.f);
        float v1 = s1[r] * inv + ((kb + 16 + nc) > qrow ? -1e9f : 0.f);
        float t = fmaxf(v0, v1);
        t = fmaxf(t, __shfl_xor(t, 1, 32));
        t = fmaxf(t, __shfl_xor(t, 2, 32));
        t = fmaxf(t, __shfl_xor(t, 4, 32));
        t = fmaxf(t, __shfl_xor(t, 8, 32));
        float mnew = fmaxf(mi[r], t);
        float sc = __expf(mi[r] - mnew);
        mi[r] = mnew;
        float p0 = __expf(v0 - mnew), p1 = __expf(v1 - mnew);
        s0[r] = p0; s1[r] = p1;
        float rs = p0 + p1;
        rs += __shfl_xor(rs, 1, 32);
        rs += __shfl_xor(rs, 2, 32);
        rs += __shfl_xor(rs, 4, 32);
        rs += __shfl_xor(rs, 8, 32);
        li[r] = li[r] * sc + rs;
#pragma unroll
        for (int j = 0; j < 4; ++j) o[j][r] *= sc;
      }
      // D-layout -> A-layout transpose of P via wave-private LDS (DS in-order)
      __bf16* Wp = Pld[wave];
#pragma unroll
      for (int r = 0; r < 8; ++r) {
        Wp[(r + 8 * half) * 32 + nc]      = (__bf16)s0[r];
        Wp[(r + 8 * half) * 32 + 16 + nc] = (__bf16)s1[r];
      }
      v16bf ap = a_frag_bf16(Wp, 32, 0, 0);
#pragma unroll
      for (int j = 0; j < 4; ++j)
        o[j] = wmma_bf16(ap, b_frag_bf16(Vt, 64, 0, j * 16), o[j]);
    }
    __syncthreads();
  }

  float* OP = out_pre + rowbase * 512 + hh * 64;
#pragma unroll
  for (int j = 0; j < 4; ++j)
#pragma unroll
    for (int r = 0; r < 8; ++r) {
      size_t m = m0l + r + 8 * half;
      OP[m * 512 + j * 16 + nc] = og * (o[j][r] / li[r]);
    }
}

// ---------------------------------------------------------------------------
// Phase 3a: z prefix per segment: zseg[t][b][hd] = 1/D + sum_{t'<t,s} sk
// ---------------------------------------------------------------------------
__global__ void zseg_kernel(const __bf16* __restrict__ skb, float* __restrict__ zseg) {
  int idx = blockIdx.x * blockDim.x + threadIdx.x;  // (b,hd), 2048 threads
  int b = idx >> 9, hd = idx & 511;
  float z = 1.0f / 64.0f;
  for (int t = 0; t < 16; ++t) {
    zseg[(t * 4 + b) * 512 + hd] = z;
    const __bf16* p = skb + ((size_t)b * 8192 + (size_t)t * 512) * 512 + hd;
    float s = 0.f;
    for (int i = 0; i < 512; ++i) s += (float)p[(size_t)i * 512];
    z += s;
  }
}

// Phase 3b: denominators denq[row]=sq.z(seg), denk[row]=sk.z(seg); 1 wave/row
__global__ void den_kernel(const __bf16* __restrict__ sqb, const __bf16* __restrict__ skb,
                           const float* __restrict__ zseg, float* __restrict__ denq,
                           float* __restrict__ denk) {
  int row = (int)((blockIdx.x * blockDim.x + threadIdx.x) >> 5);
  int l = LANE;
  int b = row >> 13, t = (row & 8191) >> 9;
  const float* z = zseg + (t * 4 + b) * 512;
  const __bf16* q = sqb + (size_t)row * 512;
  const __bf16* k = skb + (size_t)row * 512;
  float sq = 0.f, sk = 0.f;
  for (int i = l; i < 512; i += 32) {
    float zz = z[i];
    sq += (float)q[i] * zz;
    sk += (float)k[i] * zz;
  }
#pragma unroll
  for (int m = 1; m < 32; m <<= 1) {
    sq += __shfl_xor(sq, m, 32);
    sk += __shfl_xor(sk, m, 32);
  }
  if (l == 0) { denq[row] = sq; denk[row] = sk; }
}

// ---------------------------------------------------------------------------
// Phase 4a: R = [sq;sk] @ M ; q-side -> out_pre += g*R/denq,
//           k-side -> delta = V - R/denk  (M=4096, N=512, K=512).
// Memory-matrix tile staged via TDM.
// ---------------------------------------------------------------------------
__global__ void scan_read_kernel(const __bf16* __restrict__ sqb, const __bf16* __restrict__ skb,
                                 const __bf16* __restrict__ Vb, const float* __restrict__ mem,
                                 const float* __restrict__ denq, const float* __restrict__ denk,
                                 const float* __restrict__ beta, float* __restrict__ out_pre,
                                 __bf16* __restrict__ deltab, int t) {
  __shared__ __align__(16) float Mt[32 * 128];
  int wave = threadIdx.x >> 5, wm = wave >> 2, wn = wave & 3;
  int mt0 = blockIdx.x * 64 + wm * 32;
  int nblk = blockIdx.y * 128, n0l = wn * 32;
  bool kside = mt0 >= 2048;
  int mloc = kside ? (mt0 - 2048) : mt0;
  const __bf16* S = kside ? skb : sqb;
  v8f acc[2][2] = {};
  for (int k = 0; k < 512; k += 32) {
    stage_tile_f32(Mt, mem + (size_t)k * 512 + nblk, 512, 128, 32);
    v16bf a0 = a_frag_seg(S, t, mloc, k);
    v16bf a1 = a_frag_seg(S, t, mloc + 16, k);
    v16bf b0 = b_frag_f32(Mt, 128, 0, n0l);
    v16bf b1 = b_frag_f32(Mt, 128, 0, n0l + 16);
    acc[0][0] = wmma_bf16(a0, b0, acc[0][0]);
    acc[0][1] = wmma_bf16(a0, b1, acc[0][1]);
    acc[1][0] = wmma_bf16(a1, b0, acc[1][0]);
    acc[1][1] = wmma_bf16(a1, b1, acc[1][1]);
    __syncthreads();
  }
  int l = LANE, h = l >> 4, nc = l & 15;
#pragma unroll
  for (int i = 0; i < 2; ++i)
#pragma unroll
    for (int j = 0; j < 2; ++j)
#pragma unroll
      for (int r = 0; r < 8; ++r) {
        int ml = mloc + i * 16 + r + 8 * h;
        int n = nblk + n0l + j * 16 + nc;
        size_t grow = (size_t)(ml >> 9) * 8192 + (size_t)t * 512 + (ml & 511);
        float a = acc[i][j][r];
        if (!kside) {
          float gg = 1.f / (1.f + __expf(-beta[n >> 6]));
          out_pre[grow * 512 + n] += gg * (a / denq[grow]);
        } else {
          float d = (float)Vb[grow * 512 + n] - a / denk[grow];
          deltab[(size_t)ml * 512 + n] = (__bf16)d;
        }
      }
}

// Phase 4b: M += sk^T @ delta  (M=512, N=512, K=2048); LDS-staged bf16 tiles
__global__ void scan_update_kernel(const __bf16* __restrict__ skb,
                                   const __bf16* __restrict__ deltab,
                                   float* __restrict__ mem, int t) {
  __shared__ __align__(16) __bf16 At[32 * 64];   // [kk][m]
  __shared__ __align__(16) __bf16 Bt[32 * 128];  // [kk][n]
  int wave = threadIdx.x >> 5, wm = wave >> 2, wn = wave & 3;
  int mblk = blockIdx.x * 64, nblk = blockIdx.y * 128;
  int m0l = wm * 32, n0l = wn * 32;
  v8f acc[2][2] = {};
  for (int k = 0; k < 2048; k += 32) {
    {  // stage A: rows kk=k+r of sk^T source, cols mblk..mblk+63 (one uint4/thread)
      int r = threadIdx.x >> 3, c8 = (threadIdx.x & 7) * 8;
      int kk = k + r;
      size_t grow = (size_t)(kk >> 9) * 8192 + (size_t)t * 512 + (kk & 511);
      *(uint4*)(At + r * 64 + c8) = *(const uint4*)(skb + grow * 512 + mblk + c8);
      // stage B: delta rows k..k+31, cols nblk..nblk+127 (two uint4/thread)
#pragma unroll
      for (int u = 0; u < 2; ++u) {
        int i = threadIdx.x * 2 + u;
        int rb = i >> 4, cb = (i & 15) * 8;
        *(uint4*)(Bt + rb * 128 + cb) =
            *(const uint4*)(deltab + (size_t)(k + rb) * 512 + nblk + cb);
      }
    }
    __syncthreads();
    v16bf a0 = aT_frag_lds(At, m0l);
    v16bf a1 = aT_frag_lds(At, m0l + 16);
    v16bf b0 = b_frag_bf16(Bt, 128, 0, n0l);
    v16bf b1 = b_frag_bf16(Bt, 128, 0, n0l + 16);
    acc[0][0] = wmma_bf16(a0, b0, acc[0][0]);
    acc[0][1] = wmma_bf16(a0, b1, acc[0][1]);
    acc[1][0] = wmma_bf16(a1, b0, acc[1][0]);
    acc[1][1] = wmma_bf16(a1, b1, acc[1][1]);
    __syncthreads();
  }
  int l = LANE, h = l >> 4, nc = l & 15;
#pragma unroll
  for (int i = 0; i < 2; ++i)
#pragma unroll
    for (int j = 0; j < 2; ++j)
#pragma unroll
      for (int r = 0; r < 8; ++r) {
        size_t m = mblk + m0l + i * 16 + r + 8 * h;
        size_t n = nblk + n0l + j * 16 + nc;
        mem[m * 512 + n] += acc[i][j][r];
      }
}

// ---------------------------------------------------------------------------
// Phase 5: out = out_pre @ Wd (M=32768, N=64, K=512); Wd tile via TDM
// ---------------------------------------------------------------------------
__global__ void outproj_kernel(const float* __restrict__ out_pre,
                               const float* __restrict__ Wd, float* __restrict__ out) {
  __shared__ __align__(16) float Wt[32 * 64];
  int wave = threadIdx.x >> 5, wm = wave >> 1, wn = wave & 1;
  int m0 = blockIdx.x * 64 + wm * 32;
  int n0l = wn * 32;
  v8f acc[2][2] = {};
  for (int k = 0; k < 512; k += 32) {
    stage_tile_f32(Wt, Wd + (size_t)k * 64, 64, 64, 32);
    v16bf a0 = a_frag_f32(out_pre, 512, m0, k);
    v16bf a1 = a_frag_f32(out_pre, 512, m0 + 16, k);
    v16bf b0 = b_frag_f32(Wt, 64, 0, n0l);
    v16bf b1 = b_frag_f32(Wt, 64, 0, n0l + 16);
    acc[0][0] = wmma_bf16(a0, b0, acc[0][0]);
    acc[0][1] = wmma_bf16(a0, b1, acc[0][1]);
    acc[1][0] = wmma_bf16(a1, b0, acc[1][0]);
    acc[1][1] = wmma_bf16(a1, b1, acc[1][1]);
    __syncthreads();
  }
  int l = LANE, h = l >> 4, nc = l & 15;
#pragma unroll
  for (int i = 0; i < 2; ++i)
#pragma unroll
    for (int j = 0; j < 2; ++j)
#pragma unroll
      for (int r = 0; r < 8; ++r) {
        size_t m = m0 + i * 16 + r + 8 * h;
        out[m * 64 + n0l + j * 16 + nc] = acc[i][j][r];
      }
}

// ---------------------------------------------------------------------------
extern "C" void kernel_launch(void* const* d_in, const int* in_sizes, int n_in,
                              void* d_out, int out_size, void* d_ws, size_t ws_size,
                              hipStream_t stream) {
  const float* x    = (const float*)d_in[0];
  const float* Wq   = (const float*)d_in[1];
  const float* Wk   = (const float*)d_in[2];
  const float* Wv   = (const float*)d_in[3];
  const float* Wd   = (const float*)d_in[4];
  const float* beta = (const float*)d_in[5];
  float* out = (float*)d_out;

  const size_t NQ = (size_t)32768 * 512;  // B*L x HD
  char* w = (char*)d_ws;
  __bf16* Qb  = (__bf16*)w; w += NQ * 2;
  __bf16* Kb  = (__bf16*)w; w += NQ * 2;
  __bf16* Vb  = (__bf16*)w; w += NQ * 2;
  __bf16* sqb = (__bf16*)w; w += NQ * 2;
  __bf16* skb = (__bf16*)w; w += NQ * 2;
  float* out_pre = (float*)w; w += NQ * 4;
  float* mem     = (float*)w; w += (size_t)512 * 512 * 4;
  float* zseg    = (float*)w; w += (size_t)16 * 4 * 512 * 4;
  float* denq    = (float*)w; w += (size_t)32768 * 4;
  float* denk    = (float*)w; w += (size_t)32768 * 4;
  __bf16* deltab = (__bf16*)w; w += (size_t)2048 * 512 * 2;

  // Phase 1: projections (parallel over all segments)
  proj_kernel<<<dim3(512, 4, 3), 256, 0, stream>>>(x, Wq, Wk, Wv, Qb, Kb, Vb, sqb, skb);
  // Phase 2: causal attention (parallel over all segments)
  attn_kernel<<<dim3(4, 16, 32), 256, 0, stream>>>(Qb, Kb, Vb, beta, out_pre);
  // Phase 3: z prefix and denominators (parallel)
  zseg_kernel<<<dim3(8), 256, 0, stream>>>(skb, zseg);
  den_kernel<<<dim3(4096), 256, 0, stream>>>(sqb, skb, zseg, denq, denk);
  // memory state starts at zero every call (deterministic)
  hipMemsetAsync(mem, 0, (size_t)512 * 512 * 4, stream);
  // Phase 4: sequential scan over 16 segments
  for (int t = 0; t < 16; ++t) {
    scan_read_kernel<<<dim3(64, 4), 256, 0, stream>>>(sqb, skb, Vb, mem, denq, denk,
                                                      beta, out_pre, deltab, t);
    scan_update_kernel<<<dim3(8, 4), 256, 0, stream>>>(skb, deltab, mem, t);
  }
  // Phase 5: output projection
  outproj_kernel<<<dim3(512), 128, 0, stream>>>(out_pre, Wd, out);
}
